// MultiHeadedAttention_40535901340246
// MI455X (gfx1250) — compile-verified, matrix operand order checked by reading, not run on hardware
//
#include <hip/hip_runtime.h>

typedef __attribute__((ext_vector_type(16))) _Float16 v16h;
typedef __attribute__((ext_vector_type(8)))  _Float16 v8h;
typedef __attribute__((ext_vector_type(4)))  _Float16 v4h;
typedef __attribute__((ext_vector_type(8)))  float    v8f;

union F16x16 { v16h v; v8h h[2]; _Float16 e[16]; };

#define B_   2
#define T_   3
#define C_   64
#define HH   256
#define WW   256
#define DD   1024
#define NQ   4096      // 64*64 tokens per frame
#define NKV  12288     // 3 frames * 4096

// ---------------------------------------------------------------------------
// Kernel 1: scale/bias -> window partition -> LayerNorm -> f16 Q/K/V
// One block per (b, frame, token). 256 threads, 4 dims each (D=1024).
// Q is pre-scaled by 1/sqrt(D)=1/32. V is stored transposed (dim-major).
// ---------------------------------------------------------------------------
__global__ __launch_bounds__(256) void prep_qkv(
    const float* __restrict__ x,
    const float* __restrict__ qw, const float* __restrict__ qb,
    const float* __restrict__ kw, const float* __restrict__ kb,
    const float* __restrict__ vw, const float* __restrict__ vb,
    const float* __restrict__ lnq_g, const float* __restrict__ lnq_b,
    const float* __restrict__ lnk_g, const float* __restrict__ lnk_b,
    const float* __restrict__ lnv_g, const float* __restrict__ lnv_b,
    _Float16* __restrict__ qh, _Float16* __restrict__ kh,
    _Float16* __restrict__ vT)
{
  const int bid = blockIdx.x;
  const int b   = bid / (T_ * NQ);
  const int rem = bid % (T_ * NQ);
  const int n   = rem / NQ;
  const int ti  = rem % NQ;
  const int oh  = ti >> 6, ow = ti & 63;

  const int t  = threadIdx.x;
  const int c  = t >> 2;      // channel  (d = c*16 + ph*4 + pw)
  const int ph = t & 3;       // patch row

  const size_t xoff =
      ((((size_t)(b * T_ + n) * C_ + c) * HH + (oh * 4 + ph)) * WW + ow * 4);
  const float4 xv = *reinterpret_cast<const float4*>(x + xoff);
  float xs[4] = {xv.x, xv.y, xv.z, xv.w};

  const float kwc = kw[c], kbc = kb[c];
  const float vwc = vw[c], vbc = vb[c];
  const float qwc = qw[c], qbc = qb[c];

  float kx[4], vx[4], qx[4];
  float s[6] = {0.f, 0.f, 0.f, 0.f, 0.f, 0.f};
#pragma unroll
  for (int i = 0; i < 4; ++i) {
    kx[i] = xs[i] * kwc + kbc; s[0] += kx[i]; s[1] += kx[i] * kx[i];
    vx[i] = xs[i] * vwc + vbc; s[2] += vx[i]; s[3] += vx[i] * vx[i];
    qx[i] = xs[i] * qwc + qbc; s[4] += qx[i]; s[5] += qx[i] * qx[i];
  }

  __shared__ float red[6][8];
  const int wv = t >> 5, ln = t & 31;
#pragma unroll
  for (int i = 0; i < 6; ++i) {
    float r = s[i];
#pragma unroll
    for (int msk = 16; msk >= 1; msk >>= 1) r += __shfl_xor(r, msk, 32);
    if (ln == 0) red[i][wv] = r;
  }
  __syncthreads();
  float tot[6];
#pragma unroll
  for (int i = 0; i < 6; ++i) {
    float a = 0.f;
#pragma unroll
    for (int w2 = 0; w2 < 8; ++w2) a += red[i][w2];
    tot[i] = a;
  }

  const float inv = 1.0f / 1024.0f, eps = 1e-5f;
  const float mk  = tot[0] * inv;
  const float rk  = rsqrtf(fmaxf(tot[1] * inv - mk * mk, 0.f) + eps);
  const float mv_ = tot[2] * inv;
  const float rv  = rsqrtf(fmaxf(tot[3] * inv - mv_ * mv_, 0.f) + eps);
  const float mq  = tot[4] * inv;
  const float rq  = rsqrtf(fmaxf(tot[5] * inv - mq * mq, 0.f) + eps);

  const int d0    = t * 4;
  const int kvtok = n * NQ + ti;

  { // K row-major
    v4h o4;
#pragma unroll
    for (int i = 0; i < 4; ++i)
      o4[i] = (_Float16)((kx[i] - mk) * rk * lnk_g[d0 + i] + lnk_b[d0 + i]);
    *reinterpret_cast<v4h*>(kh + ((size_t)b * NKV + kvtok) * DD + d0) = o4;
  }
#pragma unroll
  for (int i = 0; i < 4; ++i) { // V transposed (dim-major)
    _Float16 hv =
        (_Float16)((vx[i] - mv_) * rv * lnv_g[d0 + i] + lnv_b[d0 + i]);
    vT[((size_t)b * DD + (d0 + i)) * NKV + kvtok] = hv;
  }
  if (n == 0) { // Q, pre-scaled by 1/sqrt(D)
    v4h o4;
#pragma unroll
    for (int i = 0; i < 4; ++i)
      o4[i] = (_Float16)(((qx[i] - mq) * rq * lnq_g[d0 + i] + lnq_b[d0 + i]) *
                         0.03125f);
    *reinterpret_cast<v4h*>(qh + ((size_t)b * NQ + ti) * DD + d0) = o4;
  }
}

// ---------------------------------------------------------------------------
// Kernel 2: flash attention, 32 queries (2 row-tiles) per block.
// Block = 8 waves; each wave owns a 128-wide slice of D=1024.
// Per 32-key chunk: each K fragment feeds 2 S-WMMAs (one per row-tile) and
// each V fragment feeds 2 PV-WMMAs, halving global traffic per matrix op.
// Partial scores are summed across waves through LDS; online softmax keeps
// per-row running max/sum; epilogue un-partitions + scale/bias + LeakyReLU.
// ---------------------------------------------------------------------------
__global__ __launch_bounds__(256) void attn_fwd(
    const _Float16* __restrict__ qh, const _Float16* __restrict__ kh,
    const _Float16* __restrict__ vT,
    const float* __restrict__ out_w, const float* __restrict__ out_b,
    float* __restrict__ out)
{
  const int lane  = threadIdx.x & 31;
  const int wave  = threadIdx.x >> 5;
  const int qtile = blockIdx.x & 127;       // 128 tiles of 32 queries
  const int b     = blockIdx.x >> 7;
  const int r     = lane & 15;              // row (A) / col (B) index
  const int hi    = lane >> 4;              // half-wave
  const int klo   = hi ? 8 : 0;             // 16-bit A-operand lane K offset

  __shared__ float sred[8][32][32];         // per-wave partial score tiles

  // Resident Q fragments: 2 row-tiles x wave's D-slice [wave*128, +128)
  F16x16 qf[2][4];
#pragma unroll
  for (int t2 = 0; t2 < 2; ++t2) {
    const _Float16* qbase =
        qh + ((size_t)(b * NQ + qtile * 32 + t2 * 16 + r)) * DD + wave * 128;
#pragma unroll
    for (int kt = 0; kt < 4; ++kt) {
      qf[t2][kt].h[0] = *(const v8h*)(qbase + kt * 32 + klo);
      qf[t2][kt].h[1] = *(const v8h*)(qbase + kt * 32 + 16 + klo);
    }
  }

  v8f o[2][8];
#pragma unroll
  for (int t2 = 0; t2 < 2; ++t2)
#pragma unroll
    for (int f = 0; f < 8; ++f) o[t2][f] = (v8f)(0.f);
  float m[2] = {-3.0e38f, -3.0e38f}, l[2] = {0.f, 0.f};

  float owv[8], obv[8];
#pragma unroll
  for (int f = 0; f < 8; ++f) {
    const int c = wave * 8 + f;
    owv[f] = out_w[c];
    obv[f] = out_b[c];
  }

  const _Float16* kbase = kh + (size_t)b * NKV * DD;
  const _Float16* vbase = vT + (size_t)b * DD * NKV;

  F16x16 pf[2];

  for (int kc = 0; kc < NKV; kc += 32) {
    // ---- partial S = Q * K^T over this wave's 128 dims, both row-tiles ----
    v8f s00 = (v8f)(0.f), s01 = (v8f)(0.f);   // tile0: keys 0-15 / 16-31
    v8f s10 = (v8f)(0.f), s11 = (v8f)(0.f);   // tile1
    const _Float16* krow0 = kbase + ((size_t)(kc + r)) * DD + wave * 128;
    const _Float16* krow1 = kbase + ((size_t)(kc + 16 + r)) * DD + wave * 128;
    if (kc + 32 < NKV) __builtin_prefetch(krow0 + 32 * DD, 0, 1);
#pragma unroll
    for (int kt = 0; kt < 4; ++kt) {
      F16x16 b0, b1;
      b0.h[0] = *(const v8h*)(krow0 + kt * 32 + klo);
      b0.h[1] = *(const v8h*)(krow0 + kt * 32 + 16 + klo);
      b1.h[0] = *(const v8h*)(krow1 + kt * 32 + klo);
      b1.h[1] = *(const v8h*)(krow1 + kt * 32 + 16 + klo);
      s00 = __builtin_amdgcn_wmma_f32_16x16x32_f16(false, qf[0][kt].v, false,
                                                   b0.v, (short)0, s00, false,
                                                   false);
      s01 = __builtin_amdgcn_wmma_f32_16x16x32_f16(false, qf[0][kt].v, false,
                                                   b1.v, (short)0, s01, false,
                                                   false);
      s10 = __builtin_amdgcn_wmma_f32_16x16x32_f16(false, qf[1][kt].v, false,
                                                   b0.v, (short)0, s10, false,
                                                   false);
      s11 = __builtin_amdgcn_wmma_f32_16x16x32_f16(false, qf[1][kt].v, false,
                                                   b1.v, (short)0, s11, false,
                                                   false);
    }

    // ---- cross-wave reduction of partial scores via LDS ----
    __syncthreads();                        // previous chunk's reads are done
#pragma unroll
    for (int j = 0; j < 8; ++j) {           // C-layout: VGPR j <-> row j+8*hi
      const int row = j + 8 * hi;
      sred[wave][row][r]           = s00[j];
      sred[wave][row][16 + r]      = s01[j];
      sred[wave][16 + row][r]      = s10[j];
      sred[wave][16 + row][16 + r] = s11[j];
    }
    __syncthreads();

    // ---- per row-tile: gather full S in A-layout, online softmax ----
#pragma unroll
    for (int t2 = 0; t2 < 2; ++t2) {
      float sv[16];
#pragma unroll
      for (int i = 0; i < 16; ++i) {
        const int col = (i < 8) ? (klo + i) : (16 + klo + (i - 8));
        float acc = 0.f;
#pragma unroll
        for (int w2 = 0; w2 < 8; ++w2) acc += sred[w2][t2 * 16 + r][col];
        sv[i] = acc;
      }

      float mc = sv[0];
#pragma unroll
      for (int i = 1; i < 16; ++i) mc = fmaxf(mc, sv[i]);
      mc = fmaxf(mc, __shfl_xor(mc, 16, 32));    // other half of the row
      const float mnew  = fmaxf(m[t2], mc);
      const float alpha = __expf(m[t2] - mnew);

      float rs = 0.f;
#pragma unroll
      for (int i = 0; i < 16; ++i) {
        const float p = __expf(sv[i] - mnew);
        rs += p;
        pf[t2].e[i] = (_Float16)p;
      }
      rs += __shfl_xor(rs, 16, 32);
      l[t2] = l[t2] * alpha + rs;
      m[t2] = mnew;

      // Rescale O accumulators (C-layout rows) by per-row alpha
#pragma unroll
      for (int j = 0; j < 8; ++j) {
        const float aj = __shfl(alpha, j + 8 * hi, 32);
#pragma unroll
        for (int f = 0; f < 8; ++f) o[t2][f][j] *= aj;
      }
    }

    // ---- O += P * V; each V fragment feeds both row-tiles ----
#pragma unroll
    for (int f = 0; f < 8; ++f) {
      const _Float16* vrow =
          vbase + ((size_t)(wave * 128 + f * 16 + r)) * NKV + kc;
      F16x16 bf;
      bf.h[0] = *(const v8h*)(vrow + klo);
      bf.h[1] = *(const v8h*)(vrow + 16 + klo);
      o[0][f] = __builtin_amdgcn_wmma_f32_16x16x32_f16(false, pf[0].v, false,
                                                       bf.v, (short)0, o[0][f],
                                                       false, false);
      o[1][f] = __builtin_amdgcn_wmma_f32_16x16x32_f16(false, pf[1].v, false,
                                                       bf.v, (short)0, o[1][f],
                                                       false, false);
    }
  }

  // ---- epilogue: 1/l, un-partition, out scale/bias, LeakyReLU ----
#pragma unroll
  for (int t2 = 0; t2 < 2; ++t2) {
    const float linv = 1.0f / l[t2];
#pragma unroll
    for (int j = 0; j < 8; ++j) {
      const float lj = __shfl(linv, j + 8 * hi, 32);
      const int q  = qtile * 32 + t2 * 16 + j + 8 * hi;
      const int oh = q >> 6, ow = q & 63;
      const int hr = oh * 4 + (r >> 2);     // ph = (d%16)/4 = r/4
      const int wc = ow * 4 + (r & 3);      // pw = d%4
#pragma unroll
      for (int f = 0; f < 8; ++f) {
        const int c = wave * 8 + f;         // d = wave*128 + f*16 + r
        float val = o[t2][f][j] * lj * owv[f] + obv[f];
        val = (val >= 0.f) ? val : 0.2f * val;
        out[(((size_t)(b * C_ + c)) * HH + hr) * WW + wc] = val;
      }
    }
  }
}

// ---------------------------------------------------------------------------
extern "C" void kernel_launch(void* const* d_in, const int* in_sizes, int n_in,
                              void* d_out, int out_size, void* d_ws,
                              size_t ws_size, hipStream_t stream) {
  (void)in_sizes; (void)n_in; (void)out_size; (void)ws_size;
  const float* x     = (const float*)d_in[0];
  const float* qw    = (const float*)d_in[1];
  const float* qb    = (const float*)d_in[2];
  const float* kw    = (const float*)d_in[3];
  const float* kb    = (const float*)d_in[4];
  const float* vw    = (const float*)d_in[5];
  const float* vb    = (const float*)d_in[6];
  const float* ow    = (const float*)d_in[7];
  const float* ob    = (const float*)d_in[8];
  const float* lnq_g = (const float*)d_in[9];
  const float* lnq_b = (const float*)d_in[10];
  const float* lnk_g = (const float*)d_in[11];
  const float* lnk_b = (const float*)d_in[12];
  const float* lnv_g = (const float*)d_in[13];
  const float* lnv_b = (const float*)d_in[14];

  _Float16* qh = (_Float16*)d_ws;                       // 2*4096*1024  f16
  _Float16* kh = qh + (size_t)B_ * NQ * DD;             // 2*12288*1024 f16
  _Float16* vT = kh + (size_t)B_ * NKV * DD;            // 2*1024*12288 f16

  prep_qkv<<<B_ * T_ * NQ, 256, 0, stream>>>(
      x, qw, qb, kw, kb, vw, vb, lnq_g, lnq_b, lnk_g, lnk_b, lnv_g, lnv_b,
      qh, kh, vT);

  attn_fwd<<<B_ * (NQ / 32), 256, 0, stream>>>(qh, kh, vT, ow, ob,
                                               (float*)d_out);
}
